// SparseAttModule_89309549953687
// MI455X (gfx1250) — matrix operand order repaired and checked by reading, not compiled, not run on hardware
//
#include <hip/hip_runtime.h>
#include <hip/hip_bf16.h>

// ---------------------------------------------------------------------------
// MI455X (gfx1250): bf16x3 split-precision WMMA GEMMs + async-to-LDS gather.
//   - all matmuls on v_wmma_f32_16x16x32_bf16 (f32 accumulate)
//   - f32 operands split into bf16 hi+lo; acc += Ah*Bh + Ah*Bl + Al*Bh
//   - conv3 gather double-buffered via GLOBAL_LOAD_ASYNC_TO_LDS_B128
//     (ASYNCcnt), prefetching neighbor kk+1 under the kk WMMA stream
// ---------------------------------------------------------------------------

#define NPTS   120000
#define NFEAT  128
#define HFEAT  64
#define TILES  (NPTS / 16)   // 7500

typedef __attribute__((ext_vector_type(16))) __bf16 v16bf;
typedef __attribute__((ext_vector_type(8)))  float  v8f;

union FragU { v16bf v; unsigned u[8]; };

#define HAS_ASYNC_LDS __has_builtin(__builtin_amdgcn_global_load_async_to_lds_b128)

#if HAS_ASYNC_LDS
typedef int async_v4i __attribute__((vector_size(16)));
typedef __attribute__((address_space(1))) async_v4i* gptr4;
typedef __attribute__((address_space(3))) async_v4i* lptr4;
__device__ __forceinline__ void async_cp16(const void* g, void* l) {
  __builtin_amdgcn_global_load_async_to_lds_b128((gptr4)(void*)g, (lptr4)l, 0, 0);
}
__device__ __forceinline__ void async_wait0() {
#if __has_builtin(__builtin_amdgcn_s_wait_asynccnt)
  __builtin_amdgcn_s_wait_asynccnt(0);
#else
  asm volatile("s_wait_asynccnt 0" ::: "memory");
#endif
}
#else
__device__ __forceinline__ void async_wait0() {}
#endif

__device__ __forceinline__ unsigned short f32_to_bf16_rne(float f) {
  unsigned u = __float_as_uint(f);
  unsigned r = (u + 0x7FFFu + ((u >> 16) & 1u)) >> 16;
  return (unsigned short)r;
}
__device__ __forceinline__ float bf16_to_f32(unsigned short h) {
  return __uint_as_float(((unsigned)h) << 16);
}
__device__ __forceinline__ void split_bf16(float x, unsigned short& hi, unsigned short& lo) {
  hi = f32_to_bf16_rne(x);
  lo = f32_to_bf16_rne(x - bf16_to_f32(hi));
}

// A fragment (16x32 bf16, MxK): lane(0-15)=row, half selects K-subblock.
// VGPR j: K = k0 + (j>=4?16:0) + half*8 + 2*(j&3), pair packed lo16=even K.
__device__ __forceinline__ v16bf load_a_frag(const unsigned short* p, int ld, int k0, int lane) {
  int half = lane >> 4, row = lane & 15;
  const unsigned short* rp = p + row * ld + k0 + half * 8;
  FragU f;
#pragma unroll
  for (int j = 0; j < 4; ++j) f.u[j]     = *(const unsigned*)(rp + 2 * j);
#pragma unroll
  for (int j = 0; j < 4; ++j) f.u[4 + j] = *(const unsigned*)(rp + 16 + 2 * j);
  return f.v;
}

// B fragment (32x16 bf16, KxN) from packed-pair weights pb[(k/2)*N + n]:
// lane(0-15)=col; VGPR j: K = (j>=4?16:0) + half*8 + 2*(j&3), pair lo16=even K.
__device__ __forceinline__ v16bf load_b_frag(const unsigned* pb, int N, int k0, int n0, int lane) {
  int half = lane >> 4, col = lane & 15;
  const unsigned* bp = pb + (size_t)(k0 / 2 + half * 4) * N + n0 + col;
  FragU f;
#pragma unroll
  for (int j = 0; j < 8; ++j) f.u[j] = bp[(size_t)(j + (j >= 4 ? 4 : 0)) * N];
  return f.v;
}

__device__ __forceinline__ v8f mma3(v8f acc, v16bf ah, v16bf al, v16bf bh, v16bf bl) {
  acc = __builtin_amdgcn_wmma_f32_16x16x32_bf16(false, ah, false, bh, (short)0, acc, false, false);
  acc = __builtin_amdgcn_wmma_f32_16x16x32_bf16(false, ah, false, bl, (short)0, acc, false, false);
  acc = __builtin_amdgcn_wmma_f32_16x16x32_bf16(false, al, false, bh, (short)0, acc, false, false);
  return acc;
}

// --- zero a small region (dummy row for invalid neighbors) -----------------
__global__ void k_zero(unsigned* __restrict__ p) { p[threadIdx.x] = 0u; }

// --- repack f32 weights [M matrices of KxN] into packed-pair hi/lo planes ---
__global__ void k_pack(const float* __restrict__ w, int K, int N, int total_pairs,
                       unsigned* __restrict__ ph, unsigned* __restrict__ pl) {
  int e = blockIdx.x * blockDim.x + threadIdx.x;
  if (e >= total_pairs) return;
  int ppm = (K / 2) * N;
  int m = e / ppm, r = e % ppm;
  int p = r / N, n = r % N;
  const float* base = w + (size_t)m * K * N;
  float a = base[(size_t)(2 * p) * N + n];
  float b = base[(size_t)(2 * p + 1) * N + n];
  unsigned short ah, al, bh, bl;
  split_bf16(a, ah, al);
  split_bf16(b, bh, bl);
  ph[e] = (unsigned)ah | ((unsigned)bh << 16);
  pl[e] = (unsigned)al | ((unsigned)bl << 16);
}

// --- t1 = relu(h @ W1)  [NPTS,128] x [128,64] -> bf16 hi/lo planes ---------
__global__ __launch_bounds__(128) void k_fc1(const float* __restrict__ hin,
                                             const unsigned* __restrict__ w_hi,
                                             const unsigned* __restrict__ w_lo,
                                             unsigned short* __restrict__ t1_hi,
                                             unsigned short* __restrict__ t1_lo) {
  __shared__ __align__(16) unsigned short ash[16 * 128];
  __shared__ __align__(16) unsigned short asl[16 * 128];
  int base = blockIdx.x * 16;
  int t = threadIdx.x;
  const float* src = hin + (size_t)base * NFEAT;
#pragma unroll
  for (int i = 0; i < 16; ++i) {           // 128 thr * 16 = 2048 elems
    int e = t * 16 + i;
    unsigned short h, l;
    split_bf16(src[e], h, l);
    ash[e] = h; asl[e] = l;
  }
  __syncthreads();
  int wave = t >> 5, lane = t & 31, n0 = wave * 16;
  int half = lane >> 4, lr = lane & 15;
  v8f acc = {0.f, 0.f, 0.f, 0.f, 0.f, 0.f, 0.f, 0.f};
#pragma unroll
  for (int k0 = 0; k0 < 128; k0 += 32) {
    v16bf Ah = load_a_frag(ash, 128, k0, lane);
    v16bf Al = load_a_frag(asl, 128, k0, lane);
    v16bf Bh = load_b_frag(w_hi, HFEAT, k0, n0, lane);
    v16bf Bl = load_b_frag(w_lo, HFEAT, k0, n0, lane);
    acc = mma3(acc, Ah, Al, Bh, Bl);
  }
#pragma unroll
  for (int v = 0; v < 8; ++v) {
    float x = acc[v];
    x = x > 0.f ? x : 0.f;
    unsigned short h, l;
    split_bf16(x, h, l);
    size_t o = (size_t)(base + v + half * 8) * HFEAT + n0 + lr;
    t1_hi[o] = h; t1_lo[o] = l;
  }
}

// stage one neighbor's 16 gathered rows into an LDS buffer (async if present)
__device__ __forceinline__ void stage_nbr(const int* __restrict__ nbr, size_t rowbase, int kk,
                                          const unsigned short* __restrict__ t1h,
                                          const unsigned short* __restrict__ t1l,
                                          const unsigned short* __restrict__ zrow,
                                          unsigned short* dsth, unsigned short* dstl,
                                          int r, int seg) {
  int idx = nbr[rowbase + kk];
  const unsigned short* sh = (idx >= 0) ? (t1h + (size_t)idx * HFEAT + seg * 8) : zrow;
  const unsigned short* sl = (idx >= 0) ? (t1l + (size_t)idx * HFEAT + seg * 8) : zrow;
#if HAS_ASYNC_LDS
  async_cp16(sh, dsth + r * HFEAT + seg * 8);
  async_cp16(sl, dstl + r * HFEAT + seg * 8);
#else
  *(uint4*)(dsth + r * HFEAT + seg * 8) = *(const uint4*)sh;
  *(uint4*)(dstl + r * HFEAT + seg * 8) = *(const uint4*)sl;
#endif
}

// --- t2 = relu(sum_k gather(t1,nbr_k) @ Wk[k]) ------------------------------
__global__ __launch_bounds__(128) void k_conv3(const unsigned short* __restrict__ t1_hi,
                                               const unsigned short* __restrict__ t1_lo,
                                               const unsigned* __restrict__ wk_hi,
                                               const unsigned* __restrict__ wk_lo,
                                               const int* __restrict__ nbr,
                                               const unsigned short* __restrict__ zrow,
                                               unsigned short* __restrict__ t2_hi,
                                               unsigned short* __restrict__ t2_lo) {
  __shared__ __align__(16) unsigned short gh[2][16 * HFEAT];
  __shared__ __align__(16) unsigned short gl[2][16 * HFEAT];
  int base = blockIdx.x * 16;
  int t = threadIdx.x, wave = t >> 5, lane = t & 31, n0 = wave * 16;
  int half = lane >> 4, lr = lane & 15;
  int r = t >> 3, seg = t & 7;             // 16 rows x 8 segments of 16B
  const size_t rowbase = (size_t)(base + r) * 27;
  v8f acc = {0.f, 0.f, 0.f, 0.f, 0.f, 0.f, 0.f, 0.f};

  stage_nbr(nbr, rowbase, 0, t1_hi, t1_lo, zrow, gh[0], gl[0], r, seg);

  for (int kk = 0; kk < 27; ++kk) {
    int cur = kk & 1;
    async_wait0();                         // my buf[cur] loads done
    __syncthreads();                       // everyone's done; prev compute done
    if (kk < 26)                           // prefetch kk+1 under the WMMAs
      stage_nbr(nbr, rowbase, kk + 1, t1_hi, t1_lo, zrow, gh[cur ^ 1], gl[cur ^ 1], r, seg);
    const unsigned* bh = wk_hi + (size_t)kk * 2048;   // 64*64/2 pairs
    const unsigned* bl = wk_lo + (size_t)kk * 2048;
#pragma unroll
    for (int k0 = 0; k0 < 64; k0 += 32) {
      v16bf Ah = load_a_frag(gh[cur], HFEAT, k0, lane);
      v16bf Al = load_a_frag(gl[cur], HFEAT, k0, lane);
      v16bf Bh = load_b_frag(bh, HFEAT, k0, n0, lane);
      v16bf Bl = load_b_frag(bl, HFEAT, k0, n0, lane);
      acc = mma3(acc, Ah, Al, Bh, Bl);
    }
  }
#pragma unroll
  for (int v = 0; v < 8; ++v) {
    float x = acc[v];
    x = x > 0.f ? x : 0.f;
    unsigned short h, l;
    split_bf16(x, h, l);
    size_t o = (size_t)(base + v + half * 8) * HFEAT + n0 + lr;
    t2_hi[o] = h; t2_lo[o] = l;
  }
}

// --- h_out = relu(t2 @ W2 + h_res)  (residual preloaded into accumulator) --
__global__ __launch_bounds__(256) void k_fc2(const unsigned short* __restrict__ t2_hi,
                                             const unsigned short* __restrict__ t2_lo,
                                             const unsigned* __restrict__ w_hi,
                                             const unsigned* __restrict__ w_lo,
                                             const float* hres, float* hout) {
  int base = blockIdx.x * 16;
  int t = threadIdx.x, wave = t >> 5, lane = t & 31, n0 = wave * 16;
  int half = lane >> 4, lr = lane & 15;
  v8f acc;
#pragma unroll
  for (int v = 0; v < 8; ++v)
    acc[v] = hres[(size_t)(base + v + half * 8) * NFEAT + n0 + lr];
  const unsigned short* ah = t2_hi + (size_t)base * HFEAT;
  const unsigned short* al = t2_lo + (size_t)base * HFEAT;
#pragma unroll
  for (int k0 = 0; k0 < 64; k0 += 32) {
    v16bf Ah = load_a_frag(ah, HFEAT, k0, lane);
    v16bf Al = load_a_frag(al, HFEAT, k0, lane);
    v16bf Bh = load_b_frag(w_hi, NFEAT, k0, n0, lane);
    v16bf Bl = load_b_frag(w_lo, NFEAT, k0, n0, lane);
    acc = mma3(acc, Ah, Al, Bh, Bl);
  }
#pragma unroll
  for (int v = 0; v < 8; ++v) {
    float x = acc[v];
    hout[(size_t)(base + v + half * 8) * NFEAT + n0 + lr] = x > 0.f ? x : 0.f;
  }
}

// --- out = a * sigmoid(b @ Wf) + x -----------------------------------------
__global__ __launch_bounds__(256) void k_final(const float* __restrict__ bbuf,
                                               const unsigned* __restrict__ wf_hi,
                                               const unsigned* __restrict__ wf_lo,
                                               const float* abuf, const float* __restrict__ xin,
                                               float* out) {
  __shared__ __align__(16) unsigned short ash[16 * 128];
  __shared__ __align__(16) unsigned short asl[16 * 128];
  int base = blockIdx.x * 16;
  int t = threadIdx.x;
  const float* src = bbuf + (size_t)base * NFEAT;
#pragma unroll
  for (int i = 0; i < 8; ++i) {            // 256 thr * 8 = 2048 elems
    int e = t * 8 + i;
    unsigned short h, l;
    split_bf16(src[e], h, l);
    ash[e] = h; asl[e] = l;
  }
  __syncthreads();
  int wave = t >> 5, lane = t & 31, n0 = wave * 16;
  int half = lane >> 4, lr = lane & 15;
  v8f acc = {0.f, 0.f, 0.f, 0.f, 0.f, 0.f, 0.f, 0.f};
#pragma unroll
  for (int k0 = 0; k0 < 128; k0 += 32) {
    v16bf Ah = load_a_frag(ash, 128, k0, lane);
    v16bf Al = load_a_frag(asl, 128, k0, lane);
    v16bf Bh = load_b_frag(wf_hi, NFEAT, k0, n0, lane);
    v16bf Bl = load_b_frag(wf_lo, NFEAT, k0, n0, lane);
    acc = mma3(acc, Ah, Al, Bh, Bl);
  }
#pragma unroll
  for (int v = 0; v < 8; ++v) {
    size_t o = (size_t)(base + v + half * 8) * NFEAT + n0 + lr;
    float s = 1.f / (1.f + __expf(-acc[v]));
    out[o] = abuf[o] * s + xin[o];
  }
}

extern "C" void kernel_launch(void* const* d_in, const int* in_sizes, int n_in,
                              void* d_out, int out_size, void* d_ws, size_t ws_size,
                              hipStream_t stream) {
  const float* x   = (const float*)d_in[0];
  const float* W1s = (const float*)d_in[1];
  const float* Wks = (const float*)d_in[2];
  const float* W2s = (const float*)d_in[3];
  const float* Wf  = (const float*)d_in[4];
  const int*   nbr = (const int*)d_in[5];
  float* out = (float*)d_out;

  char* ws = (char*)d_ws;
  size_t off = 0;
  auto alloc = [&](size_t bytes) -> void* {
    void* p = ws + off;
    off = (off + bytes + 255) & ~(size_t)255;
    return p;
  };

  const int P1 = 6 * 128 * 64 / 2;         // 24576 packed pairs
  const int PK = 6 * 27 * 64 * 64 / 2;     // 331776
  const int P2 = 6 * 64 * 128 / 2;         // 24576
  const int PF = 128 * 128 / 2;            // 8192

  unsigned* w1h = (unsigned*)alloc((size_t)P1 * 4);
  unsigned* w1l = (unsigned*)alloc((size_t)P1 * 4);
  unsigned* wkh = (unsigned*)alloc((size_t)PK * 4);
  unsigned* wkl = (unsigned*)alloc((size_t)PK * 4);
  unsigned* w2h = (unsigned*)alloc((size_t)P2 * 4);
  unsigned* w2l = (unsigned*)alloc((size_t)P2 * 4);
  unsigned* wfh = (unsigned*)alloc((size_t)PF * 4);
  unsigned* wfl = (unsigned*)alloc((size_t)PF * 4);
  float* hB = (float*)alloc((size_t)NPTS * NFEAT * 4);
  unsigned short* t1h = (unsigned short*)alloc((size_t)NPTS * HFEAT * 2);
  unsigned short* t1l = (unsigned short*)alloc((size_t)NPTS * HFEAT * 2);
  unsigned short* t2h = (unsigned short*)alloc((size_t)NPTS * HFEAT * 2);
  unsigned short* t2l = (unsigned short*)alloc((size_t)NPTS * HFEAT * 2);
  unsigned short* zrow = (unsigned short*)alloc(256);   // zero dummy row

  k_zero<<<1, 64, 0, stream>>>((unsigned*)zrow);
  k_pack<<<(P1 + 255) / 256, 256, 0, stream>>>(W1s, 128, 64, P1, w1h, w1l);
  k_pack<<<(PK + 255) / 256, 256, 0, stream>>>(Wks, 64, 64, PK, wkh, wkl);
  k_pack<<<(P2 + 255) / 256, 256, 0, stream>>>(W2s, 64, 128, P2, w2h, w2l);
  k_pack<<<(PF + 255) / 256, 256, 0, stream>>>(Wf, 128, 128, PF, wfh, wfl);

  // branch a: units 0..2, working buffer = d_out
  for (int i = 0; i < 3; ++i) {
    const float* hin = (i == 0) ? x : out;
    k_fc1<<<TILES, 128, 0, stream>>>(hin, w1h + i * 4096, w1l + i * 4096, t1h, t1l);
    k_conv3<<<TILES, 128, 0, stream>>>(t1h, t1l, wkh + (size_t)i * 27 * 2048,
                                       wkl + (size_t)i * 27 * 2048, nbr, zrow, t2h, t2l);
    k_fc2<<<TILES, 256, 0, stream>>>(t2h, t2l, w2h + i * 4096, w2l + i * 4096, hin, out);
  }
  // branch b: units 3..5, working buffer = hB
  for (int i = 3; i < 6; ++i) {
    const float* hin = (i == 3) ? x : hB;
    k_fc1<<<TILES, 128, 0, stream>>>(hin, w1h + i * 4096, w1l + i * 4096, t1h, t1l);
    k_conv3<<<TILES, 128, 0, stream>>>(t1h, t1l, wkh + (size_t)i * 27 * 2048,
                                       wkl + (size_t)i * 27 * 2048, nbr, zrow, t2h, t2l);
    k_fc2<<<TILES, 256, 0, stream>>>(t2h, t2l, w2h + i * 4096, w2l + i * 4096, hin, hB);
  }
  k_final<<<TILES, 256, 0, stream>>>(hB, wfh, wfl, out, x, out);
}
